// HOE_1_2_Module_63041529970916
// MI455X (gfx1250) — matrix operand order repaired and checked by reading, not compiled, not run on hardware
//
#include <hip/hip_runtime.h>
#include <hip/hip_bf16.h>

typedef __attribute__((ext_vector_type(16))) _Float16 v16h;
typedef __attribute__((ext_vector_type(8)))  float    v8f;
typedef __attribute__((ext_vector_type(2)))  float    v2f;

#define BS 4
#define UE 8
#define RXN 128
#define DD 64
#define NBU (BS*UE)           // 32

// ---------------------------------------------------------------------------
// Kernel 1: per (b,u) compute
//   Row[i][d]  = x[i] @ W1b^T + m @ W1e^T + b1      (f-block [64:128], [256:320])
//   Col[j][d]  = x[j] @ W1c^T                       (f-block [128:192])
//   Diag[i][d] = x[i] @ W1a^T + m @ W1d^T           (f-block [0:64],  [192:256])
// via V_WMMA_F32_16X16X4_F32 (full fp32). Block 0 also converts W2 -> f16.
// ---------------------------------------------------------------------------
__global__ __launch_bounds__(256) void hoe_phase1(
    const float* __restrict__ x,    // [NBU][RXN][DD]
    const float* __restrict__ W1,   // [DD][5*DD]
    const float* __restrict__ b1,   // [DD]
    const float* __restrict__ W2,   // [DD][DD]
    float* __restrict__ Row, float* __restrict__ Col, float* __restrict__ Diag,
    _Float16* __restrict__ W2h)
{
    __shared__ float lds_part[4][DD];
    __shared__ float lds_m[DD];
    __shared__ float lds_rowm[DD];
    __shared__ float lds_diagm[DD];

    const int bu  = blockIdx.x;      // 0..31
    const int tid = threadIdx.x;     // 0..255

    if (bu == 0) {
        for (int e = tid; e < DD * DD; e += 256)
            W2h[e] = (_Float16)W2[e];
    }

    // ---- mean over rx (128 rows) ----
    const float* xb = x + (size_t)bu * RXN * DD;
    {
        const int d = tid & 63, part = tid >> 6;
        float s = 0.f;
        const int r0 = part * 32;
        #pragma unroll 8
        for (int r = 0; r < 32; ++r) s += xb[(r0 + r) * DD + d];
        lds_part[part][d] = s;
    }
    __syncthreads();
    if (tid < DD)
        lds_m[tid] = (lds_part[0][tid] + lds_part[1][tid] +
                      lds_part[2][tid] + lds_part[3][tid]) * (1.f / 128.f);
    __syncthreads();
    if (tid < DD) {
        float acc = b1[tid];
        #pragma unroll 8
        for (int k = 0; k < DD; ++k) acc += lds_m[k] * W1[tid * 320 + 256 + k];
        lds_rowm[tid] = acc;
    } else if (tid < 2 * DD) {
        const int d = tid - DD;
        float acc = 0.f;
        #pragma unroll 8
        for (int k = 0; k < DD; ++k) acc += lds_m[k] * W1[d * 320 + 192 + k];
        lds_diagm[d] = acc;
    }
    __syncthreads();

    // ---- WMMA f32 16x16x4: wave w handles i-tile w*16, all 3 targets ----
    const int wave = tid >> 5, lane = tid & 31;
    const int hi = lane >> 4, lm = lane & 15;
    const int i0 = wave * 16;

    // A fragments for full K=64: 16 k-steps of K=4
    const float* xrow = xb + (size_t)(i0 + lm) * DD;
    v2f a[16];
    #pragma unroll
    for (int ks = 0; ks < 16; ++ks)
        a[ks] = *(const v2f*)&xrow[ks * 4 + 2 * hi];

    float*     outs [3] = { Diag, Row, Col };
    const int  fbase[3] = { 0,    64,  128 };

    #pragma unroll
    for (int t = 0; t < 3; ++t) {
        float* op = outs[t] + (size_t)bu * RXN * DD;
        #pragma unroll
        for (int nt = 0; nt < 4; ++nt) {
            const int n = nt * 16 + lm;
            const float cinit = (t == 0) ? lds_diagm[n] : (t == 1) ? lds_rowm[n] : 0.f;
            v8f c = { cinit, cinit, cinit, cinit, cinit, cinit, cinit, cinit };
            const float* wrow = W1 + (size_t)n * 320 + fbase[t];
            #pragma unroll
            for (int ks = 0; ks < 16; ++ks) {
                v2f bfrag = *(const v2f*)&wrow[ks * 4 + 2 * hi];
                c = __builtin_amdgcn_wmma_f32_16x16x4_f32(
                        false, a[ks], false, bfrag, (short)0, c, false, false);
            }
            #pragma unroll
            for (int r = 0; r < 8; ++r)
                op[(size_t)(i0 + r + hi * 8) * DD + n] = c[r];
        }
    }
}

// ---------------------------------------------------------------------------
// Kernel 2: per wave: (bu, i, j-tile of 16).
//   h[j][d] = Row[i][d] + Col[j][d] + (i==j)*Diag[i][d]
//   LayerNorm(d), ReLU, +bias on diagonal, then y @ W2^T + b2 via
//   V_WMMA_F32_16X16X32_F16 (2 chained K-chunks), direct global store.
// Lane pair (m, m+16) owns pair m; lane's d-set d = 16q + 8*hi + r matches the
// 16-bit A-fragment VGPR layout exactly (no cross-lane packing needed).
// ---------------------------------------------------------------------------
__global__ __launch_bounds__(256) void hoe_phase2(
    const float* __restrict__ Row, const float* __restrict__ Col,
    const float* __restrict__ Diag, const _Float16* __restrict__ W2h,
    const float* __restrict__ ln_g, const float* __restrict__ ln_b,
    const float* __restrict__ bias, const float* __restrict__ b2,
    float* __restrict__ out)
{
    const int tid  = threadIdx.x;
    const int g    = blockIdx.x * 8 + (tid >> 5);   // global wave id, 0..32767
    const int bu   = g >> 10;
    const int rem  = g & 1023;
    const int i    = rem >> 3;
    const int jt   = rem & 7;

    const int lane = tid & 31;
    const int hi   = lane >> 4;       // which half-row of K this lane holds
    const int m    = lane & 15;       // pair index within j-tile
    const int j    = jt * 16 + m;
    const bool isdiag = ((i >> 4) == jt) && (m == (i & 15));

    const float* rowp = Row  + (size_t)(bu * RXN + i) * DD;
    const float* colp = Col  + (size_t)(bu * RXN + j) * DD;
    const float* dgp  = Diag + (size_t)(bu * RXN + i) * DD;

    float h[4][8];
    float s1 = 0.f, s2 = 0.f;
    #pragma unroll
    for (int q = 0; q < 4; ++q) {
        const int d0 = q * 16 + hi * 8;
        float4 ra = *(const float4*)&rowp[d0];
        float4 rb = *(const float4*)&rowp[d0 + 4];
        float4 ca = *(const float4*)&colp[d0];
        float4 cb = *(const float4*)&colp[d0 + 4];
        h[q][0] = ra.x + ca.x; h[q][1] = ra.y + ca.y;
        h[q][2] = ra.z + ca.z; h[q][3] = ra.w + ca.w;
        h[q][4] = rb.x + cb.x; h[q][5] = rb.y + cb.y;
        h[q][6] = rb.z + cb.z; h[q][7] = rb.w + cb.w;
        if (isdiag) {
            float4 da = *(const float4*)&dgp[d0];
            float4 db = *(const float4*)&dgp[d0 + 4];
            h[q][0] += da.x; h[q][1] += da.y; h[q][2] += da.z; h[q][3] += da.w;
            h[q][4] += db.x; h[q][5] += db.y; h[q][6] += db.z; h[q][7] += db.w;
        }
        #pragma unroll
        for (int r = 0; r < 8; ++r) { s1 += h[q][r]; s2 += h[q][r] * h[q][r]; }
    }
    // reduce across the lane pair (lane ^ 16) to cover all 64 d
    s1 += __shfl_xor(s1, 16);
    s2 += __shfl_xor(s2, 16);
    const float mu  = s1 * (1.f / 64.f);
    const float var = s2 * (1.f / 64.f) - mu * mu;
    const float rs  = __frsqrt_rn(var + 1e-5f);

    // LN -> ReLU -> diag bias -> f16 A fragments (lane-local by construction)
    v16h a0, a1;
    #pragma unroll
    for (int q = 0; q < 4; ++q) {
        const int d0 = q * 16 + hi * 8;
        float4 ga = *(const float4*)&ln_g[d0];
        float4 gb = *(const float4*)&ln_g[d0 + 4];
        float4 ba = *(const float4*)&ln_b[d0];
        float4 bb = *(const float4*)&ln_b[d0 + 4];
        float gg[8] = { ga.x, ga.y, ga.z, ga.w, gb.x, gb.y, gb.z, gb.w };
        float bbv[8] = { ba.x, ba.y, ba.z, ba.w, bb.x, bb.y, bb.z, bb.w };
        #pragma unroll
        for (int r = 0; r < 8; ++r) {
            float y = (h[q][r] - mu) * rs * gg[r] + bbv[r];
            y = fmaxf(y, 0.f);
            if (isdiag) y += bias[d0 + r];
            const _Float16 yh = (_Float16)y;
            if (q < 2) a0[(q & 1) * 8 + r] = yh;
            else       a1[(q & 1) * 8 + r] = yh;
        }
    }

    // fc2: D[m, o] = sum_d y * W2[o][d] + b2[o]
    const size_t obase = (size_t)(bu * RXN + i) * RXN * DD + (size_t)(jt * 16) * DD;
    const int kb = hi * 8;    // B layout: lane half hi holds K 8hi..8hi+7 and 16+8hi..16+8hi+7
    typedef __attribute__((ext_vector_type(8))) _Float16 v8h_;
    #pragma unroll
    for (int nt = 0; nt < 4; ++nt) {
        const int o = nt * 16 + m;
        v16h bf0 = __builtin_shufflevector(*(const v8h_*)&W2h[o * DD + kb], *(const v8h_*)&W2h[o * DD + 16 + kb], 0,1,2,3,4,5,6,7,8,9,10,11,12,13,14,15);        // K chunk 0..31
        v16h bf1 = __builtin_shufflevector(*(const v8h_*)&W2h[o * DD + 32 + kb], *(const v8h_*)&W2h[o * DD + 48 + kb], 0,1,2,3,4,5,6,7,8,9,10,11,12,13,14,15);   // K chunk 32..63
        const float b2v = b2[o];
        v8f c = { b2v, b2v, b2v, b2v, b2v, b2v, b2v, b2v };
        c = __builtin_amdgcn_wmma_f32_16x16x32_f16(false, a0, false, bf0,
                                                   (short)0, c, false, false);
        c = __builtin_amdgcn_wmma_f32_16x16x32_f16(false, a1, false, bf1,
                                                   (short)0, c, false, false);
        #pragma unroll
        for (int r = 0; r < 8; ++r) {
            const int p = r + hi * 8;                      // pair (row) index
            out[obase + (size_t)p * DD + o] = c[r];
        }
    }
}

extern "C" void kernel_launch(void* const* d_in, const int* in_sizes, int n_in,
                              void* d_out, int out_size, void* d_ws, size_t ws_size,
                              hipStream_t stream) {
    const float* x    = (const float*)d_in[0];
    const float* W1   = (const float*)d_in[1];
    const float* b1   = (const float*)d_in[2];
    const float* ln_g = (const float*)d_in[3];
    const float* ln_b = (const float*)d_in[4];
    const float* bias = (const float*)d_in[5];
    const float* W2   = (const float*)d_in[6];
    const float* b2   = (const float*)d_in[7];
    float* out = (float*)d_out;

    float* Row  = (float*)d_ws;                       // 32*128*64 f32 = 1 MiB
    float* Col  = Row  + NBU * RXN * DD;
    float* Diag = Col  + NBU * RXN * DD;
    _Float16* W2h = (_Float16*)(Diag + NBU * RXN * DD);  // 8 KiB

    hipLaunchKernelGGL(hoe_phase1, dim3(NBU), dim3(256), 0, stream,
                       x, W1, b1, W2, Row, Col, Diag, W2h);
    hipLaunchKernelGGL(hoe_phase2, dim3(4096), dim3(256), 0, stream,
                       Row, Col, Diag, W2h, ln_g, ln_b, bias, b2, out);
}
